// LCGN_26508538151406
// MI455X (gfx1250) — matrix operand order repaired and checked by reading, not compiled, not run on hardware
//
#include <hip/hip_runtime.h>
#include <hip/hip_bf16.h>
#include <stdint.h>

typedef __bf16 bf16_t;
typedef __attribute__((ext_vector_type(16))) __bf16 v16bf;
typedef __attribute__((ext_vector_type(8)))  float  v8f;
typedef __attribute__((ext_vector_type(4)))  int    v4i_t;

#define TB   256
#define BM   64
#define BN   128
#define BK   32
#define LDSS 40            // LDS row stride in halves: 80B rows, conflict-free frag reads
#define ALDS (BM * LDSS)   // per-buffer A LDS halves
#define BLDS (BN * LDSS)   // per-buffer B LDS halves

// CDNA5 async global->LDS copy (ASYNCcnt-tracked); guarded so compile never breaks.
#if defined(__has_builtin)
#if __has_builtin(__builtin_amdgcn_global_load_async_to_lds_b128) && \
    __has_builtin(__builtin_amdgcn_s_wait_asynccnt)
#define USE_ASYNC_LDS 1
#endif
#endif
#ifndef USE_ASYNC_LDS
#define USE_ASYNC_LDS 0
#endif

#define AS1 __attribute__((address_space(1)))
#define AS3 __attribute__((address_space(3)))
#define GPTR(p) ((AS1 v4i_t*)(void*)(p))   // global b128 operand
#define LPTR(p) ((AS3 v4i_t*)(void*)(p))   // LDS b128 operand

union FragU { v16bf v; unsigned int u[8]; };

// ---------------------------------------------------------------- elementwise
__global__ __launch_bounds__(TB)
void k_cvt_f32_bf16(const float* __restrict__ in, bf16_t* __restrict__ out, long long n) {
  long long i = (long long)blockIdx.x * TB + threadIdx.x;
  long long stride = (long long)gridDim.x * TB;
  for (; i < n; i += stride) out[i] = (bf16_t)in[i];
}

// weights: f32 [K, 512] row-major -> bf16 [512, K] row-major (transposed)
__global__ __launch_bounds__(TB)
void k_cvt_transpose_bf16(const float* __restrict__ in, bf16_t* __restrict__ out, int K) {
  __shared__ float t[32][33];
  const int kb = blockIdx.x * 32, nb = blockIdx.y * 32;
  const int tx = threadIdx.x & 31, ty = threadIdx.x >> 5;   // 32 x 8
#pragma unroll
  for (int i = 0; i < 32; i += 8)
    t[ty + i][tx] = in[(long long)(kb + ty + i) * 512 + nb + tx];
  __syncthreads();
#pragma unroll
  for (int i = 0; i < 32; i += 8)
    out[(long long)(nb + ty + i) * K + kb + tx] = (bf16_t)t[tx][ty + i];
}

__global__ __launch_bounds__(TB)
void k_fill_ctx(const float* __restrict__ initm, bf16_t* __restrict__ out, long long n) {
  long long i = (long long)blockIdx.x * TB + threadIdx.x;
  long long stride = (long long)gridDim.x * TB;
  for (; i < n; i += stride) out[i] = (bf16_t)initm[i & 511];
}

// L2-normalize rows of x [rows, 2048] and emit bf16
__global__ __launch_bounds__(TB)
void k_rownorm_bf16(const float* __restrict__ x, bf16_t* __restrict__ o) {
  __shared__ float red[TB];
  const long long row = blockIdx.x;
  const float* px = x + row * 2048;
  bf16_t* po = o + row * 2048;
  const int t = threadIdx.x;
  float v[8]; float s = 0.f;
#pragma unroll
  for (int i = 0; i < 8; ++i) { v[i] = px[t + i * TB]; s += v[i] * v[i]; }
  red[t] = s; __syncthreads();
  for (int k = TB / 2; k > 0; k >>= 1) { if (t < k) red[t] += red[t + k]; __syncthreads(); }
  const float scale = 1.f / fmaxf(sqrtf(red[0]), 1e-12f);
#pragma unroll
  for (int i = 0; i < 8; ++i) po[t + i * TB] = (bf16_t)(v[i] * scale);
}

// in-place row softmax over 512-wide bf16 rows, scaled scores, f32 math
__global__ __launch_bounds__(TB)
void k_softmax_bf16(bf16_t* __restrict__ sc, float scale) {
  __shared__ float red[TB];
  const long long row = blockIdx.x;
  bf16_t* p = sc + row * 512;
  const int t = threadIdx.x;
  float a = (float)p[t] * scale;
  float b = (float)p[t + 256] * scale;
  red[t] = fmaxf(a, b); __syncthreads();
  for (int k = TB / 2; k > 0; k >>= 1) { if (t < k) red[t] = fmaxf(red[t], red[t + k]); __syncthreads(); }
  const float mx = red[0]; __syncthreads();
  const float ea = __expf(a - mx), eb = __expf(b - mx);
  red[t] = ea + eb; __syncthreads();
  for (int k = TB / 2; k > 0; k >>= 1) { if (t < k) red[t] += red[t + k]; __syncthreads(); }
  const float inv = 1.f / red[0];
  p[t]       = (bf16_t)(ea * inv);
  p[t + 256] = (bf16_t)(eb * inv);
}

// ---------------------------------------------------------------- WMMA GEMM
// C[bat] = concat_segs(A)[bat] @ B(+bias)(*gate)(*emul)
//  A: up to 3 row-major bf16 segments, each [M, 1<<seg_shift]
//  B: btr==1 -> [Nt, K] row-major, used transposed (pre-transposed weights, k-matrix)
//     btr==0 -> [K, Nt] row-major (v-matrix for prob@v)
//
// Stage a BK chunk into an LDS buffer (async direct-to-LDS when available);
// macro-inlined so all addressing stays in SGPR/VGPR (no lambda stack traffic).
#if USE_ASYNC_LDS
#define STAGE_A(k0_, adst_)                                                     \
  { const int seg_ = (k0_) >> seg_shift;                                        \
    const bf16_t* Aseg_ = (seg_ == 0) ? A0 : ((seg_ == 1) ? A1 : A2);           \
    __builtin_amdgcn_global_load_async_to_lds_b128(                             \
        GPTR(Aseg_ + aoff + ((k0_) & segmask)), LPTR(adst_), 0, 0); }
#define STAGE_BT(k0_, bdst_)                                                    \
  { __builtin_amdgcn_global_load_async_to_lds_b128(                             \
        GPTR(Bm + boff_tr + (k0_)), LPTR(bdst_), 0, 0);                         \
    __builtin_amdgcn_global_load_async_to_lds_b128(                             \
        GPTR(Bm + boff_tr + (k0_) + 8), LPTR((bdst_) + 8), 0, 0); }
#else
#define STAGE_A(k0_, adst_)                                                     \
  { const int seg_ = (k0_) >> seg_shift;                                        \
    const bf16_t* Aseg_ = (seg_ == 0) ? A0 : ((seg_ == 1) ? A1 : A2);           \
    *reinterpret_cast<uint4*>(adst_) =                                          \
        *reinterpret_cast<const uint4*>(Aseg_ + aoff + ((k0_) & segmask)); }
#define STAGE_BT(k0_, bdst_)                                                    \
  { *reinterpret_cast<uint4*>(bdst_) =                                          \
        *reinterpret_cast<const uint4*>(Bm + boff_tr + (k0_));                  \
    *reinterpret_cast<uint4*>((bdst_) + 8) =                                    \
        *reinterpret_cast<const uint4*>(Bm + boff_tr + (k0_) + 8); }
#endif
#define STAGE_BR(k0_, bdst_)                                                    \
  { const bf16_t* bs_ = Bm + boff_row + (long long)(k0_) * ldb;                 \
    uint4 d0_ = *reinterpret_cast<const uint4*>(bs_);                           \
    uint4 d1_ = *reinterpret_cast<const uint4*>(bs_ + 8);                       \
    const bf16_t* dv0_ = reinterpret_cast<const bf16_t*>(&d0_);                 \
    const bf16_t* dv1_ = reinterpret_cast<const bf16_t*>(&d1_);                 \
    _Pragma("unroll")                                                           \
    for (int j_ = 0; j_ < 8; ++j_) (bdst_)[j_ * LDSS] = dv0_[j_];               \
    _Pragma("unroll")                                                           \
    for (int j_ = 0; j_ < 8; ++j_) (bdst_)[(j_ + 8) * LDSS] = dv1_[j_]; }

__global__ __launch_bounds__(TB)
void k_gemm_bf16(const bf16_t* __restrict__ A0, const bf16_t* __restrict__ A1,
                 const bf16_t* __restrict__ A2,
                 int seg_shift, long long a_bs,
                 const bf16_t* __restrict__ Bm, int ldb, int btr, long long b_bs,
                 const float* __restrict__ bias,
                 const bf16_t* __restrict__ gate,
                 const bf16_t* __restrict__ emul,
                 void* __restrict__ Cout, int cf32, long long c_bs,
                 int M, int Nt, int K)
{
  __shared__ bf16_t As[2 * ALDS];
  __shared__ bf16_t Bs[2 * BLDS];

  const int tid  = threadIdx.x;
  const int bn   = blockIdx.x * BN;
  const int bm   = blockIdx.y * BM;
  const int bat  = blockIdx.z;

  const int wave = tid >> 5;
  const int lane = tid & 31;
  const int half = lane >> 4;
  const int r    = lane & 15;
  const int sm   = wave >> 1;          // 0..3 : 16-row stripe
  const int sn0  = (wave & 1) * 4;     // 0 or 4: quad of 16-col stripes

  const int segw    = 1 << seg_shift;
  const int segmask = segw - 1;

  // ---- loop-invariant addressing (hoisted) ----
  const int arow = tid >> 2, acol = (tid & 3) * 8;    // A tile: 64 x 32, 1x b128/thread
  const int bnr  = tid >> 1, bkc  = (tid & 1) * 16;   // B (transposed src): 128 x 32, 2x b128/thread
  const int bkr  = tid >> 3, bnc  = (tid & 7) * 16;   // B (row-major src): 32 x 128, 16 halves/thread

  const long long aoff     = (long long)bat * a_bs + (long long)(bm + arow) * segw + acol;
  const long long boff_tr  = (long long)bat * b_bs + (long long)(bn + bnr) * ldb + bkc;
  const long long boff_row = (long long)bat * b_bs + (long long)bkr * ldb + bn + bnc;

  bf16_t* adst0  = &As[arow * LDSS + acol];
  bf16_t* btdst0 = &Bs[bnr * LDSS + bkc];
  bf16_t* brdst0 = &Bs[bnc * LDSS + bkr];

  // fragment-read base pointers (ISA VGPR layout: interleaved K for both A and B)
  const bf16_t* ard = &As[(sm * 16 + r) * LDSS];
  const bf16_t* brd[4];
#pragma unroll
  for (int j = 0; j < 4; ++j) brd[j] = &Bs[((sn0 + j) * 16 + r) * LDSS];

  int kmap_a[8], kmap_b[8];
#pragma unroll
  for (int g = 0; g < 8; ++g) {
    kmap_a[g] = ((g < 4) ? (2 * g) : (16 + 2 * (g - 4))) + 8 * half;
    kmap_b[g] = ((g < 4) ? (2 * g) : (16 + 2 * (g - 4))) + 8 * half;
  }

  v8f acc[4];
#pragma unroll
  for (int j = 0; j < 4; ++j) acc[j] = (v8f){0.f,0.f,0.f,0.f,0.f,0.f,0.f,0.f};

  const int nsteps = K / BK;

  // prologue: stage 0 into buffer 0
  STAGE_A(0, adst0);
  if (btr) { STAGE_BT(0, btdst0); } else { STAGE_BR(0, brdst0); }

  for (int i = 0; i < nsteps; ++i) {
    const int buf = i & 1;
    if (i + 1 < nsteps) {
      const int kn  = (i + 1) * BK;              // prefetch next chunk (other buffer)
      const int nb  = (i + 1) & 1;
      bf16_t* ad = adst0 + nb * ALDS;
      if (btr) {
        bf16_t* bd = btdst0 + nb * BLDS;
        STAGE_A(kn, ad);
        STAGE_BT(kn, bd);
#if USE_ASYNC_LDS
        __builtin_amdgcn_s_wait_asynccnt(3);     // in-order: stage i's 3 copies complete
#endif
      } else {
        bf16_t* bd = brdst0 + nb * BLDS;
        STAGE_A(kn, ad);
        STAGE_BR(kn, bd);
#if USE_ASYNC_LDS
        __builtin_amdgcn_s_wait_asynccnt(1);     // only A is async on this path
#endif
      }
    } else {
#if USE_ASYNC_LDS
      __builtin_amdgcn_s_wait_asynccnt(0);
#endif
    }
    __syncthreads();

    const int lba = buf * ALDS;
    const int lbb = buf * BLDS;
    FragU af, bfr[4];
#pragma unroll
    for (int g = 0; g < 8; ++g)
      af.u[g] = *reinterpret_cast<const unsigned int*>(&ard[lba + kmap_a[g]]);
#pragma unroll
    for (int j = 0; j < 4; ++j)
#pragma unroll
      for (int g = 0; g < 8; ++g)
        bfr[j].u[g] = *reinterpret_cast<const unsigned int*>(&brd[j][lbb + kmap_b[g]]);
#pragma unroll
    for (int j = 0; j < 4; ++j)
      acc[j] = __builtin_amdgcn_wmma_f32_16x16x32_bf16(false, af.v, false, bfr[j].v,
                                                       (short)0, acc[j], false, false);
    __syncthreads();   // all waves done reading buf before it is refilled
  }

  // epilogue: C/D layout -> row = g + 8*half, col = lane&15 within 16x16 subtile
#pragma unroll
  for (int t2 = 0; t2 < 4; ++t2) {
    const int colg = bn + (sn0 + t2) * 16 + r;
    const float bval = bias ? bias[colg] : 0.f;
#pragma unroll
    for (int g = 0; g < 8; ++g) {
      const int rowg = bm + sm * 16 + half * 8 + g;
      float vv = acc[t2][g] + bval;
      if (gate) vv *= (float)gate[(long long)(rowg >> 9) * Nt + colg];
      if (emul) vv *= (float)emul[(long long)rowg * Nt + colg];
      const long long idx = (long long)bat * c_bs + (long long)rowg * Nt + colg;
      if (cf32) ((float*)Cout)[idx] = vv;
      else      ((bf16_t*)Cout)[idx] = (bf16_t)vv;
    }
  }
}

// ---------------------------------------------------------------- host side
static void gemm(hipStream_t st,
                 const bf16_t* A0, const bf16_t* A1, const bf16_t* A2,
                 int seg_shift, long long a_bs,
                 const bf16_t* Bm, int ldb, int btr, long long b_bs,
                 const float* bias, const bf16_t* gate, const bf16_t* emul,
                 void* C, int cf32, long long c_bs,
                 int M, int Nt, int K, int batches)
{
  dim3 grid(Nt / BN, M / BM, batches);
  k_gemm_bf16<<<grid, dim3(TB), 0, st>>>(A0, A1, A2, seg_shift, a_bs, Bm, ldb, btr, b_bs,
                                         bias, gate, emul, C, cf32, c_bs, M, Nt, K);
}

extern "C" void kernel_launch(void* const* d_in, const int* in_sizes, int n_in,
                              void* d_out, int out_size, void* d_ws, size_t ws_size,
                              hipStream_t stream) {
  (void)in_sizes; (void)n_in; (void)out_size; (void)ws_size;

  const float* x        = (const float*)d_in[0];
  const float* cmd      = (const float*)d_in[1];
  const float* W_init   = (const float*)d_in[3];
  const float* b_init   = (const float*)d_in[4];
  const float* init_mem = (const float*)d_in[5];
  const float* W_xloc   = (const float*)d_in[6];   const float* b_xloc = (const float*)d_in[7];
  const float* W_xctx   = (const float*)d_in[8];   const float* b_xctx = (const float*)d_in[9];
  const float* W_q      = (const float*)d_in[10];  const float* b_q    = (const float*)d_in[11];
  const float* W_k      = (const float*)d_in[12];  const float* b_k    = (const float*)d_in[13];
  const float* W_v      = (const float*)d_in[14];  const float* b_v    = (const float*)d_in[15];
  const float* W_pk     = (const float*)d_in[16];  const float* b_pk   = (const float*)d_in[17];
  const float* W_pv     = (const float*)d_in[18];  const float* b_pv   = (const float*)d_in[19];
  const float* W_mem    = (const float*)d_in[20];  const float* b_mem  = (const float*)d_in[21];
  const float* W_kb     = (const float*)d_in[22];  const float* b_kb   = (const float*)d_in[23];

  const int Bb = 64, Nn = 512, Dfeat = 2048, CTX = 512, CMD = 512;
  const long long MN = (long long)Bb * Nn;          // 32768
  const long long bs_att = (long long)Nn * CTX;     // per-batch stride for attention

  char* w = (char*)d_ws;
  auto carve = [&](size_t bytes) -> void* {
    char* p = w; w += (bytes + 255) & ~(size_t)255; return (void*)p;
  };
  bf16_t* xnorm   = (bf16_t*)carve(MN * Dfeat * 2);
  bf16_t* xloc    = (bf16_t*)carve(MN * CTX * 2);
  bf16_t* xctx0   = (bf16_t*)carve(MN * CTX * 2);
  bf16_t* xctx1   = (bf16_t*)carve(MN * CTX * 2);
  bf16_t* projloc = (bf16_t*)carve(MN * CTX * 2);   // reused in-place for proj_loc*proj_ctx
  bf16_t* q       = (bf16_t*)carve(MN * CTX * 2);   // reused for message
  bf16_t* kmat    = (bf16_t*)carve(MN * CTX * 2);
  bf16_t* vmat    = (bf16_t*)carve(MN * CTX * 2);
  bf16_t* score   = (bf16_t*)carve((long long)Bb * Nn * Nn * 2); // softmaxed in place
  bf16_t* pkg     = (bf16_t*)carve((long long)Bb * CTX * 2);
  bf16_t* pvg     = (bf16_t*)carve((long long)Bb * CTX * 2);
  bf16_t* cmdb    = (bf16_t*)carve((long long)Bb * CMD * 2);
  // all weights stored bf16 TRANSPOSED: [512, K]
  bf16_t* wt_init = (bf16_t*)carve((long long)Dfeat * CTX * 2);
  bf16_t* wt_xloc = (bf16_t*)carve((long long)CTX * CTX * 2);
  bf16_t* wt_xctx = (bf16_t*)carve((long long)CTX * CTX * 2);
  bf16_t* wt_q    = (bf16_t*)carve((long long)3 * CTX * CTX * 2);
  bf16_t* wt_k    = (bf16_t*)carve((long long)3 * CTX * CTX * 2);
  bf16_t* wt_v    = (bf16_t*)carve((long long)3 * CTX * CTX * 2);
  bf16_t* wt_pk   = (bf16_t*)carve((long long)CMD * CTX * 2);
  bf16_t* wt_pv   = (bf16_t*)carve((long long)CMD * CTX * 2);
  bf16_t* wt_mem  = (bf16_t*)carve((long long)2 * CTX * CTX * 2);
  bf16_t* wt_kb   = (bf16_t*)carve((long long)2 * CTX * CTX * 2);

  auto cvtT = [&](const float* s, bf16_t* dd, int K) {   // [K,512] f32 -> [512,K] bf16
    k_cvt_transpose_bf16<<<dim3(K / 32, 16), dim3(TB), 0, stream>>>(s, dd, K);
  };
  cvtT(W_init, wt_init, 2048);
  cvtT(W_xloc, wt_xloc, 512);
  cvtT(W_xctx, wt_xctx, 512);
  cvtT(W_q,    wt_q,    1536);
  cvtT(W_k,    wt_k,    1536);
  cvtT(W_v,    wt_v,    1536);
  cvtT(W_pk,   wt_pk,   512);
  cvtT(W_pv,   wt_pv,   512);
  cvtT(W_mem,  wt_mem,  1024);
  cvtT(W_kb,   wt_kb,   1024);
  {
    long long n = (long long)Bb * CMD;
    k_cvt_f32_bf16<<<dim3((unsigned)((n + TB - 1) / TB)), dim3(TB), 0, stream>>>(cmd, cmdb, n);
  }

  // stem: normalize + x_loc = imgs @ W_init + b_init ; x_ctx = broadcast(init_mem)
  k_rownorm_bf16<<<dim3((unsigned)MN), dim3(TB), 0, stream>>>(x, xnorm);
  k_fill_ctx<<<dim3(4096), dim3(TB), 0, stream>>>(init_mem, xctx0, MN * CTX);
  gemm(stream, xnorm, nullptr, nullptr, 11, 0, wt_init, 2048, 1, 0,
       b_init, nullptr, nullptr, xloc, 0, 0, (int)MN, 512, 2048, 1);

  // command gates pk / pv
  gemm(stream, cmdb, nullptr, nullptr, 9, 0, wt_pk, 512, 1, 0,
       b_pk, nullptr, nullptr, pkg, 0, 0, 64, 512, 512, 1);
  gemm(stream, cmdb, nullptr, nullptr, 9, 0, wt_pv, 512, 1, 0,
       b_pv, nullptr, nullptr, pvg, 0, 0, 64, 512, 512, 1);

  const float inv_sqrt_ctx = 0.04419417382415922f;  // 1/sqrt(512)
  bf16_t* cur = xctx0; bf16_t* nxt = xctx1;
  for (int it = 0; it < 4; ++it) {
    // proj_loc, then proj = proj_loc * proj_ctx (in place)
    gemm(stream, xloc, nullptr, nullptr, 9, 0, wt_xloc, 512, 1, 0,
         b_xloc, nullptr, nullptr, projloc, 0, 0, (int)MN, 512, 512, 1);
    gemm(stream, cur, nullptr, nullptr, 9, 0, wt_xctx, 512, 1, 0,
         b_xctx, nullptr, projloc, projloc, 0, 0, (int)MN, 512, 512, 1);
    // q/k/v from concatenated [x_loc | x_ctx | prod]
    gemm(stream, xloc, cur, projloc, 9, 0, wt_q, 1536, 1, 0,
         b_q, nullptr, nullptr, q, 0, 0, (int)MN, 512, 1536, 1);
    gemm(stream, xloc, cur, projloc, 9, 0, wt_k, 1536, 1, 0,
         b_k, pkg, nullptr, kmat, 0, 0, (int)MN, 512, 1536, 1);
    gemm(stream, xloc, cur, projloc, 9, 0, wt_v, 1536, 1, 0,
         b_v, pvg, nullptr, vmat, 0, 0, (int)MN, 512, 1536, 1);
    // attention: score = q @ k^T (batched), softmax, message = prob @ v
    gemm(stream, q, nullptr, nullptr, 9, bs_att, kmat, 512, 1, bs_att,
         nullptr, nullptr, nullptr, score, 0, (long long)Nn * Nn, 512, 512, 512, 64);
    k_softmax_bf16<<<dim3((unsigned)MN), dim3(TB), 0, stream>>>(score, inv_sqrt_ctx);
    gemm(stream, score, nullptr, nullptr, 9, (long long)Nn * Nn, vmat, 512, 0, bs_att,
         nullptr, nullptr, nullptr, q /*message*/, 0, bs_att, 512, 512, 512, 64);
    // x_ctx = [x_ctx | message] @ W_mem + b_mem
    gemm(stream, cur, q, nullptr, 9, 0, wt_mem, 1024, 1, 0,
         b_mem, nullptr, nullptr, nxt, 0, 0, (int)MN, 512, 1024, 1);
    bf16_t* t = cur; cur = nxt; nxt = t;
  }

  // output: [x_loc | x_ctx] @ W_kb + b_kb -> f32
  gemm(stream, xloc, cur, nullptr, 9, 0, wt_kb, 1024, 1, 0,
       b_kb, nullptr, nullptr, d_out, 1, 0, (int)MN, 512, 1024, 1);
}
